// CrossAttention_58342835749158
// MI455X (gfx1250) — matrix operand order repaired and checked by reading, not compiled, not run on hardware
//
#include <hip/hip_runtime.h>
#include <stdint.h>

// ---------------------------------------------------------------------------
// CrossAttention block for MI455X (gfx1250, wave32, WMMA 16x16x32 bf16).
// B=4, C=256, N=4096, C_cond=512, M=1024, heads=8, d=32. ~40 GFLOP vs ~50MB
// minimal HBM traffic @23.3TB/s -> compute bound -> bf16 WMMA everywhere.
// Producers write activations in WMMA-fragment-ready layouts so consumers do
// pure b128 loads straight into v_wmma operands. Attention K/V chunks are
// DMA'd into LDS double buffers with the Tensor Data Mover (TENSORcnt path),
// removing the 4x redundant per-wave global streams within a block.
// ---------------------------------------------------------------------------

typedef __attribute__((ext_vector_type(16))) __bf16 v16bf;
typedef __attribute__((ext_vector_type(8)))  float  v8f;
typedef uint32_t u32x4 __attribute__((ext_vector_type(4)));
typedef int      i32x4 __attribute__((ext_vector_type(4)));
typedef int      i32x8 __attribute__((ext_vector_type(8)));

static constexpr int BATCH  = 4;
static constexpr int C_IN   = 256;
static constexpr int NQ     = 4096;
static constexpr int C_CND  = 512;
static constexpr int NKV    = 1024;
static constexpr int NHEAD  = 8;
static constexpr int DHEAD  = 32;
static constexpr int HIDDEN = 1024;

#if defined(__has_builtin)
#if __has_builtin(__builtin_amdgcn_tensor_load_to_lds) && \
    __has_builtin(__builtin_amdgcn_s_wait_tensorcnt)
#define USE_TDM 1
#endif
#endif

__device__ __forceinline__ uint16_t f2bf_u(float f) {
  union { float f; uint32_t u; } v; v.f = f;
  uint32_t r = v.u + 0x7FFFu + ((v.u >> 16) & 1u);   // round-to-nearest-even
  return (uint16_t)(r >> 16);
}

// B-matrix fragment-ready layout ("bswz"):
// per 32(K)x16(N) tile: lane = ((k%16)/8)*16 + n%16 owns halfword 8*((k%32)/16) + (k&7).
// One fragment == 32 contiguous bytes per lane == two b128 loads.
__device__ __forceinline__ size_t bswz_idx(int bb, int k, int n, int KT, int NT) {
  int kt = k >> 5, kr = k & 31;
  size_t tile = ((size_t)bb * NT + (n >> 4)) * KT + kt;
  return (tile * 32 + (((kr >> 3) & 1) << 4) + (n & 15)) * 16 + ((kr >> 4) << 3) + (kr & 7);
}

// A-matrix fragment-ready layout ("aswz") for K-dim = 32 (Q tiles):
// inverse of: halfword h=2j+parity holds K = 16*(j>>2) + 8*laneHi + 2*(j&3) + parity
__device__ __forceinline__ size_t aswz_idx(int bh, int k, int n, int NT) {
  int parity = k & 1, t = k >> 1;
  int laneHi = (t >> 2) & 1;
  int j = ((t >> 3) << 2) + (t & 3);
  size_t tile = (size_t)bh * NT + (n >> 4);
  return (tile * 32 + laneHi * 16 + (n & 15)) * 16 + j * 2 + parity;
}

// XOR butterfly via ds_swizzle (group-of-32 mode: and=0x1f, or=0, xor=m).
#define SWZ_XOR(v, m) \
  __int_as_float(__builtin_amdgcn_ds_swizzle(__float_as_int(v), ((m) << 10) | 0x1f))

__device__ __forceinline__ float redmax16(float v) {
  v = fmaxf(v, SWZ_XOR(v, 1));
  v = fmaxf(v, SWZ_XOR(v, 2));
  v = fmaxf(v, SWZ_XOR(v, 4));
  v = fmaxf(v, SWZ_XOR(v, 8));
  return v;
}
__device__ __forceinline__ float redsum16(float v) {
  v += SWZ_XOR(v, 1);
  v += SWZ_XOR(v, 2);
  v += SWZ_XOR(v, 4);
  v += SWZ_XOR(v, 8);
  return v;
}

#ifdef USE_TDM
// DMA (rows x row_units-of-8B) global -> LDS via the Tensor Data Mover.
// rows==0 -> 1D copy (tile_dim1 unused). Descriptor per CDNA5 ISA 8.3-8.6.
__device__ __forceinline__ void tdm_load(uint32_t lds_off, const void* gptr,
                                         int row_units, int rows,
                                         int row_stride_units) {
  uint64_t ga = (uint64_t)(uintptr_t)gptr;
  u32x4 g0;
  g0[0] = 1u;                                    // count=1, user descriptor
  g0[1] = lds_off;                               // LDS byte address
  g0[2] = (uint32_t)ga;                          // global addr [31:0]
  g0[3] = (uint32_t)(ga >> 32) | (2u << 30);     // global addr [56:32] | type=2
  i32x8 g1;
  g1[0] = 3 << 16;                               // data_size = 3 (8 bytes)
  g1[1] = 0;                                     // no atomic barrier
  g1[2] = 16;                                    // tensor_dim0 = 1<<20
  g1[3] = 16 | (row_units << 16);                // tensor_dim1 = 1<<20 | tile_dim0
  g1[4] = rows & 0xffff;                         // tile_dim1 (0 => 1D)
  g1[5] = row_stride_units;                      // tensor_dim0_stride [31:0]
  g1[6] = 0;
  g1[7] = 0;
  i32x4 gz; gz[0] = 0; gz[1] = 0; gz[2] = 0; gz[3] = 0;
#if __clang_major__ >= 23
  i32x8 gz8; gz8[0]=0; gz8[1]=0; gz8[2]=0; gz8[3]=0; gz8[4]=0; gz8[5]=0; gz8[6]=0; gz8[7]=0;
  __builtin_amdgcn_tensor_load_to_lds(g0, g1, gz, gz, gz8, 0);
#else
  __builtin_amdgcn_tensor_load_to_lds(g0, g1, gz, gz, 0);
#endif
}
#endif

// --------------------- converts (weights linear, cond bswz) -----------------
__global__ void cvt_k(const float* __restrict__ in, uint16_t* __restrict__ out,
                      int n, float scale) {
  int i = blockIdx.x * 256 + threadIdx.x;
  if (i < n) out[i] = f2bf_u(in[i] * scale);
}

__global__ void cvt_cond_k(const float* __restrict__ in, uint16_t* __restrict__ out) {
  int i = blockIdx.x * 256 + threadIdx.x;          // B*512*1024 = 2^21 elements
  int b = i >> 19;
  int rem = i & 524287;
  int k = rem >> 10, n = rem & 1023;
  out[bswz_idx(b, k, n, C_CND >> 5, NKV >> 4)] = f2bf_u(in[i]);
}

// ------------------------------- GroupNorm ---------------------------------
// One block per (batch, group): 8 channels x 4096 px. Output bswz bf16.
__global__ __launch_bounds__(256) void gn_k(
    const float* __restrict__ x, const float* __restrict__ w,
    const float* __restrict__ bia, uint16_t* __restrict__ out) {
  __shared__ float rs[256], rss[256];
  const int g = blockIdx.x & 31;
  const int b = blockIdx.x >> 5;
  const size_t base = ((size_t)b * C_IN + (size_t)g * 8) * NQ;
  const int cnt = 8 * NQ;
  float s = 0.f, ss = 0.f;
  for (int i = threadIdx.x; i < cnt; i += 256) {
    float v = x[base + i]; s += v; ss += v * v;
  }
  rs[threadIdx.x] = s; rss[threadIdx.x] = ss;
  __syncthreads();
  for (int off = 128; off > 0; off >>= 1) {
    if ((int)threadIdx.x < off) {
      rs[threadIdx.x]  += rs[threadIdx.x + off];
      rss[threadIdx.x] += rss[threadIdx.x + off];
    }
    __syncthreads();
  }
  float mean = rs[0] / (float)cnt;
  float var  = rss[0] / (float)cnt - mean * mean;
  float rinv = rsqrtf(var + 1e-5f);
  for (int i = threadIdx.x; i < cnt; i += 256) {
    int c = g * 8 + (i >> 12);
    int n = i & 4095;
    float v = (x[base + i] - mean) * rinv * w[c] + bia[c];
    out[bswz_idx(b, c, n, C_IN >> 5, NQ >> 4)] = f2bf_u(v);
  }
}

// ------------------------------ WMMA GEMM ----------------------------------
// D[b][CO][NN] = W[CO][CK] * X[b][CK][NN]; W row-major bf16, X in bswz.
// 256 thr = 8 waves; wave tile 16(O) x 64(N); inner loop: 10 b128 + 4 wmma.
enum { OUT_F32 = 0, OUT_BSWZ = 1, OUT_QSWZ = 2, OUT_KSWZ = 3, OUT_VSWZ = 4 };

template <int MODE, bool BIAS, bool GELU, bool RES>
__global__ __launch_bounds__(256, 2) void gemm_k(
    const uint16_t* __restrict__ Wb, const uint16_t* __restrict__ Xb,
    const float* __restrict__ bias, const float* __restrict__ res,
    void* __restrict__ outp, int CO, int CK, int NN) {
  const int b     = blockIdx.z;
  const int nbase = blockIdx.x * 64;
  const int wid   = threadIdx.x >> 5;
  const int lane  = threadIdx.x & 31;
  const int m     = lane & 15;
  const int hi    = lane >> 4;
  const int ka    = hi * 8;
  const int obase = blockIdx.y * 128 + wid * 16;
  const int KT = CK >> 5, NT = NN >> 4;

  v8f acc[4];
  for (int i = 0; i < 4; ++i)
    for (int r = 0; r < 8; ++r) acc[i][r] = 0.f;

  const uint16_t* wrow = Wb + (size_t)(obase + m) * CK + ka;
  const uint16_t* xcol = Xb + (((size_t)b * NT + (nbase >> 4)) * KT * 32 + lane) * 16;

  for (int kt = 0; kt < KT; ++kt) {
    union { v16bf v; uint4 q[2]; } Af;
    Af.q[0] = *(const uint4*)(wrow + kt * 32);
    Af.q[1] = *(const uint4*)(wrow + kt * 32 + 16);
    const uint16_t* xk = xcol + (size_t)kt * 512;
#pragma unroll
    for (int nt = 0; nt < 4; ++nt) {
      union { v16bf v; uint4 q[2]; } Bf;
      const uint16_t* p = xk + (size_t)nt * KT * 512;
      Bf.q[0] = *(const uint4*)(p);
      Bf.q[1] = *(const uint4*)(p + 8);
      acc[nt] = __builtin_amdgcn_wmma_f32_16x16x32_bf16(
          false, Af.v, false, Bf.v, (short)0, acc[nt], false, false);
    }
  }

#pragma unroll
  for (int nt = 0; nt < 4; ++nt) {
    int n = nbase + nt * 16 + m;
#pragma unroll
    for (int r = 0; r < 8; ++r) {
      int o = obase + r + hi * 8;           // C layout: vgpr r -> rows r / r+8
      float v = acc[nt][r];
      if (BIAS) v += bias[o];
      if (GELU) v = 0.5f * v * (1.f + erff(v * 0.70710678118f));
      if (RES)  v += res[(size_t)b * CO * NN + (size_t)o * NN + n];
      if (MODE == OUT_F32) {
        ((float*)outp)[(size_t)b * CO * NN + (size_t)o * NN + n] = v;
      } else {
        uint16_t hv = f2bf_u(v);
        uint16_t* ou = (uint16_t*)outp;
        if (MODE == OUT_BSWZ)
          ou[bswz_idx(b, o, n, CO >> 5, NN >> 4)] = hv;
        else if (MODE == OUT_QSWZ)
          ou[aswz_idx(b * NHEAD + (o >> 5), o & 31, n, NN >> 4)] = hv;
        else if (MODE == OUT_KSWZ)
          ou[bswz_idx(b * NHEAD + (o >> 5), o & 31, n, 1, NN >> 4)] = hv;
        else  // OUT_VSWZ: per (b,h) matrix [keys=NN][d=32] as B-operand
          ou[bswz_idx(b * NHEAD + (o >> 5), n, o & 31, NN >> 5, 2)] = hv;
      }
    }
  }
}

// --------------------------- Flash attention -------------------------------
// Grid (NQ/64, heads, batch), 128 thr = 4 waves, wave = 16 queries.
// 64-key chunks: 4 score WMMAs + 4 PV WMMAs per chunk. K/V chunks are TDM-
// DMA'd into LDS double buffers (wave0 issues, all waves consume); softmax
// reductions amortized over 64 keys; scale folded into wq upstream.
__global__ __launch_bounds__(128, 2) void attn_k(
    const uint16_t* __restrict__ Qsw, const uint16_t* __restrict__ Ksw,
    const uint16_t* __restrict__ Vsw, uint16_t* __restrict__ Ob) {
  __shared__ __align__(16) uint16_t pl[4][16 * 64];
  __shared__ __align__(16) uint16_t Kls[2][4 * 512];   // 4 key-tiles / chunk
  __shared__ __align__(16) uint16_t Vls[2][2 * 1024];  // 2 d-rows x 2 tiles
  const int b = blockIdx.z, h = blockIdx.y, bh = b * NHEAD + h;
  const int wid  = threadIdx.x >> 5;
  const int lane = threadIdx.x & 31;
  const int m = lane & 15, hi = lane >> 4;
  const int ka = hi * 8;
  const int qt = blockIdx.x * 4 + wid;      // 16-query tile index
  uint16_t* pw = pl[wid];

  // Q fragment (pre-scaled by 1/16 via wq): two b128 loads, kernel-resident.
  union { v16bf v; uint4 q[2]; } Af;
  const uint16_t* qp = Qsw + (((size_t)bh * (NQ >> 4) + qt) * 32 + lane) * 16;
  Af.q[0] = *(const uint4*)(qp);
  Af.q[1] = *(const uint4*)(qp + 8);

  // block-uniform global bases for K/V chunk DMA
  const uint16_t* kgb = Ksw + (size_t)bh * (NKV >> 4) * 512;
  const uint16_t* vgb = Vsw + (size_t)bh * 2 * (NKV >> 5) * 512;

  v8f acc0, acc1, zero;
  float mrun[8], lrun[8];
  for (int r = 0; r < 8; ++r) {
    acc0[r] = 0.f; acc1[r] = 0.f; zero[r] = 0.f;
    mrun[r] = -1e30f; lrun[r] = 0.f;
  }

#ifdef USE_TDM
  if (wid == 0) {   // prologue: DMA chunk 0 (K: 1D 4KB; V: 2 rows x 2KB)
    tdm_load((uint32_t)(uintptr_t)&Kls[0][0], kgb, 512, 0, 512);
    tdm_load((uint32_t)(uintptr_t)&Vls[0][0], vgb, 256, 2, 4096);
  }
  __builtin_amdgcn_s_wait_tensorcnt(0);
  __syncthreads();
#endif

  for (int c = 0; c < (NKV >> 6); ++c) {    // 16 chunks of 64 keys
#ifdef USE_TDM
    const int buf = c & 1;
    if (wid == 0 && c + 1 < (NKV >> 6)) {   // DMA next chunk into other buffer
      tdm_load((uint32_t)(uintptr_t)&Kls[buf ^ 1][0],
               kgb + (size_t)(4 * (c + 1)) * 512, 512, 0, 512);
      tdm_load((uint32_t)(uintptr_t)&Vls[buf ^ 1][0],
               vgb + (size_t)(2 * (c + 1)) * 512, 256, 2, 4096);
    }
#else
    const int buf = 0;
    __syncthreads();
    for (int i = threadIdx.x; i < 512; i += 128) {   // cooperative copy (uint4)
      if (i < 256)
        *(uint4*)&Kls[0][i * 8] = *(const uint4*)(kgb + (size_t)(4 * c) * 512 + i * 8);
      else {
        int vi = i - 256, nt = vi >> 7, w = vi & 127;
        *(uint4*)&Vls[0][nt * 1024 + w * 8] =
            *(const uint4*)(vgb + (size_t)(2 * c) * 512 + (size_t)nt * 16384 + w * 8);
      }
    }
    __syncthreads();
#endif

    // S = Q * K  (four 16-key tiles; each B-frag is two b128 LDS loads)
    v8f s[4];
#pragma unroll
    for (int nt = 0; nt < 4; ++nt) {
      union { v16bf v; uint4 q[2]; } Bf;
      const uint16_t* p = &Kls[buf][nt * 512 + lane * 16];
      Bf.q[0] = *(const uint4*)(p);
      Bf.q[1] = *(const uint4*)(p + 8);
      s[nt] = __builtin_amdgcn_wmma_f32_16x16x32_bf16(
          false, Af.v, false, Bf.v, (short)0, zero, false, false);
    }
    // online softmax over the 64 keys of this chunk
#pragma unroll
    for (int r = 0; r < 8; ++r) {
      float t0 = s[0][r], t1 = s[1][r], t2 = s[2][r], t3 = s[3][r];
      float nm = fmaxf(mrun[r],
                       redmax16(fmaxf(fmaxf(t0, t1), fmaxf(t2, t3))));
      float p0 = __expf(t0 - nm), p1 = __expf(t1 - nm);
      float p2 = __expf(t2 - nm), p3 = __expf(t3 - nm);
      float rsum = redsum16((p0 + p1) + (p2 + p3));
      float corr = __expf(mrun[r] - nm);
      lrun[r] = lrun[r] * corr + rsum;
      mrun[r] = nm;
      acc0[r] *= corr;
      acc1[r] *= corr;
      int row = r + hi * 8;                  // C-layout -> row-major P in LDS
      pw[row * 64 + m]      = f2bf_u(p0);
      pw[row * 64 + m + 16] = f2bf_u(p1);
      pw[row * 64 + m + 32] = f2bf_u(p2);
      pw[row * 64 + m + 48] = f2bf_u(p3);
    }
    // restripe P (16 queries x 64 keys) into two A fragments: 4 b128 LDS loads
    union { v16bf v; uint4 q[2]; } Pa0, Pa1;
    const uint16_t* pr = pw + m * 64 + ka;
    Pa0.q[0] = *(const uint4*)(pr);
    Pa0.q[1] = *(const uint4*)(pr + 16);
    Pa1.q[0] = *(const uint4*)(pr + 32);
    Pa1.q[1] = *(const uint4*)(pr + 48);
    // O += P * V  (two 32-key sub-chunks x two 16-wide d tiles)
#pragma unroll
    for (int nt = 0; nt < 2; ++nt) {
      union { v16bf v; uint4 q[2]; } Vf0, Vf1;
      const uint16_t* p0 = &Vls[buf][nt * 1024 + lane * 16];
      const uint16_t* p1 = p0 + 512;
      Vf0.q[0] = *(const uint4*)(p0);
      Vf0.q[1] = *(const uint4*)(p0 + 8);
      Vf1.q[0] = *(const uint4*)(p1);
      Vf1.q[1] = *(const uint4*)(p1 + 8);
      if (nt == 0) {
        acc0 = __builtin_amdgcn_wmma_f32_16x16x32_bf16(
            false, Pa0.v, false, Vf0.v, (short)0, acc0, false, false);
        acc0 = __builtin_amdgcn_wmma_f32_16x16x32_bf16(
            false, Pa1.v, false, Vf1.v, (short)0, acc0, false, false);
      } else {
        acc1 = __builtin_amdgcn_wmma_f32_16x16x32_bf16(
            false, Pa0.v, false, Vf0.v, (short)0, acc1, false, false);
        acc1 = __builtin_amdgcn_wmma_f32_16x16x32_bf16(
            false, Pa1.v, false, Vf1.v, (short)0, acc1, false, false);
      }
    }
#ifdef USE_TDM
    __builtin_amdgcn_s_wait_tensorcnt(0);   // next chunk arrived (issuer wave)
    __syncthreads();                        // + everyone done with buf
#endif
  }

  // normalize + store attention output in bswz (consumed by wo projection)
#pragma unroll
  for (int r = 0; r < 8; ++r) {
    float inv = 1.f / lrun[r];
    int ln = r + hi * 8;                     // q & 15
    size_t base0 = (((size_t)(b * (NQ >> 4) + qt) * (C_IN >> 5) + h) * 32) * 16;
    Ob[base0 + (size_t)((m >> 3) * 16 + ln) * 16 + 0 * 8 + (m & 7)] = f2bf_u(acc0[r] * inv);
    Ob[base0 + (size_t)((m >> 3) * 16 + ln) * 16 + 1 * 8 + (m & 7)] = f2bf_u(acc1[r] * inv);
  }
}

// ------------------------------- launcher ----------------------------------
extern "C" void kernel_launch(void* const* d_in, const int* in_sizes, int n_in,
                              void* d_out, int out_size, void* d_ws, size_t ws_size,
                              hipStream_t stream) {
  (void)in_sizes; (void)n_in; (void)out_size; (void)ws_size;
  const float* input  = (const float*)d_in[0];
  const float* cond   = (const float*)d_in[1];
  const float* norm_w = (const float*)d_in[2];
  const float* norm_b = (const float*)d_in[3];
  const float* wq     = (const float*)d_in[4];
  const float* wk     = (const float*)d_in[5];
  const float* wv     = (const float*)d_in[6];
  const float* wo     = (const float*)d_in[7];
  const float* bo     = (const float*)d_in[8];
  const float* n2w    = (const float*)d_in[9];
  const float* n2b    = (const float*)d_in[10];
  const float* m1w    = (const float*)d_in[11];
  const float* m1b    = (const float*)d_in[12];
  const float* m2w    = (const float*)d_in[13];
  const float* m2b    = (const float*)d_in[14];

  char* ws = (char*)d_ws;
  size_t off = 0;
  auto alloc = [&](size_t bytes) -> char* {
    char* p = ws + off;
    off += (bytes + 255) & ~(size_t)255;
    return p;
  };
  uint16_t* WQb = (uint16_t*)alloc((size_t)C_IN * C_IN * 2);
  uint16_t* WKb = (uint16_t*)alloc((size_t)C_IN * C_CND * 2);
  uint16_t* WVb = (uint16_t*)alloc((size_t)C_IN * C_CND * 2);
  uint16_t* WOb = (uint16_t*)alloc((size_t)C_IN * C_IN * 2);
  uint16_t* W1b = (uint16_t*)alloc((size_t)HIDDEN * C_IN * 2);
  uint16_t* W2b = (uint16_t*)alloc((size_t)C_IN * HIDDEN * 2);
  uint16_t* CNb = (uint16_t*)alloc((size_t)BATCH * C_CND * NKV * 2);   // bswz
  uint16_t* NRb = (uint16_t*)alloc((size_t)BATCH * C_IN * NQ * 2);     // bswz
  uint16_t* Qsw = (uint16_t*)alloc((size_t)BATCH * C_IN * NQ * 2);     // aswz/bh
  uint16_t* Ksw = (uint16_t*)alloc((size_t)BATCH * C_IN * NKV * 2);    // bswz/bh
  uint16_t* Vsw = (uint16_t*)alloc((size_t)BATCH * NKV * C_IN * 2);    // bswz/bh
  uint16_t* Atb = (uint16_t*)alloc((size_t)BATCH * C_IN * NQ * 2);     // bswz
  float*    ORs = (float*)   alloc((size_t)BATCH * C_IN * NQ * 4);
  uint16_t* Y2b = (uint16_t*)alloc((size_t)BATCH * C_IN * NQ * 2);     // bswz
  uint16_t* Hb  = (uint16_t*)alloc((size_t)BATCH * HIDDEN * NQ * 2);   // bswz

  auto cvt = [&](const float* src, uint16_t* dst, int n, float scale) {
    cvt_k<<<(n + 255) / 256, 256, 0, stream>>>(src, dst, n, scale);
  };
  cvt(wq,  WQb, C_IN * C_IN, 0.0625f);     // fold softmax 1/sqrt(C) into wq
  cvt(wk,  WKb, C_IN * C_CND, 1.f);
  cvt(wv,  WVb, C_IN * C_CND, 1.f);
  cvt(wo,  WOb, C_IN * C_IN, 1.f);
  cvt(m1w, W1b, HIDDEN * C_IN, 1.f);
  cvt(m2w, W2b, C_IN * HIDDEN, 1.f);
  cvt_cond_k<<<(BATCH * C_CND * NKV) / 256, 256, 0, stream>>>(cond, CNb);

  // GroupNorm 1 -> bswz bf16
  gn_k<<<dim3(BATCH * 32), 256, 0, stream>>>(input, norm_w, norm_b, NRb);

  // q/k/v projections straight into attention-fragment layouts
  gemm_k<OUT_QSWZ, false, false, false>
      <<<dim3(NQ / 64, C_IN / 128, BATCH), 256, 0, stream>>>(
          WQb, NRb, nullptr, nullptr, Qsw, C_IN, C_IN, NQ);
  gemm_k<OUT_KSWZ, false, false, false>
      <<<dim3(NKV / 64, C_IN / 128, BATCH), 256, 0, stream>>>(
          WKb, CNb, nullptr, nullptr, Ksw, C_IN, C_CND, NKV);
  gemm_k<OUT_VSWZ, false, false, false>
      <<<dim3(NKV / 64, C_IN / 128, BATCH), 256, 0, stream>>>(
          WVb, CNb, nullptr, nullptr, Vsw, C_IN, C_CND, NKV);

  // flash attention -> bswz
  attn_k<<<dim3(NQ / 64, NHEAD, BATCH), 128, 0, stream>>>(Qsw, Ksw, Vsw, Atb);

  // out = wo @ attn + bo + input  (fp32, kept for residual + GN2)
  gemm_k<OUT_F32, true, false, true>
      <<<dim3(NQ / 64, C_IN / 128, BATCH), 256, 0, stream>>>(
          WOb, Atb, bo, input, ORs, C_IN, C_IN, NQ);

  // GroupNorm 2 -> bswz bf16
  gn_k<<<dim3(BATCH * 32), 256, 0, stream>>>(ORs, n2w, n2b, Y2b);

  // mlp1 + exact GELU -> bswz hidden
  gemm_k<OUT_BSWZ, true, true, false>
      <<<dim3(NQ / 64, HIDDEN / 128, BATCH), 256, 0, stream>>>(
          W1b, Y2b, m1b, nullptr, Hb, HIDDEN, C_IN, NQ);

  // mlp2 + bias + residual -> fp32 output
  gemm_k<OUT_F32, true, false, true>
      <<<dim3(NQ / 64, C_IN / 128, BATCH), 256, 0, stream>>>(
          W2b, Hb, m2b, ORs, d_out, C_IN, HIDDEN, NQ);
}
